// encoder_89721866814270
// MI455X (gfx1250) — matrix operand order repaired and checked by reading, not compiled, not run on hardware
//
#include <hip/hip_runtime.h>
#include <hip/hip_bf16.h>

// ---------------------------------------------------------------------------
// Problem sizes (compile-time, from reference): B=256, D=256, L=256, H=512
// ---------------------------------------------------------------------------
#define Bn 256
#define Dn 256
#define Ln 256
#define Hn 512
#define KTOT 768              // D + H : fused K for gates GEMM
#define NG   2048             // 4*H gate columns
#define KCH  (KTOT / 32)      // 24 k-chunks per step

typedef __attribute__((ext_vector_type(16))) __bf16 v16bf;
typedef __attribute__((ext_vector_type(8)))  float  v8f;

union Frag16 { uint4 q[2]; v16bf v; };

// ---------------------------------------------------------------------------
// Workspace layout (bytes)
// ---------------------------------------------------------------------------
#define OFF_ATTN   0u                         // B*D f32          = 256 KB
#define OFF_WCAT   (256u*1024u)               // 2048*768 bf16    = 3 MB
#define OFF_WBF    (OFF_WCAT + 3u*1024u*1024u)// L*B*D bf16       = 32 MB
#define OFF_HBUF   (OFF_WBF + 32u*1024u*1024u)// 2*B*H bf16       = 512 KB
#define OFF_BAR    (OFF_HBUF + 512u*1024u)    // barrier counter

// ---------------------------------------------------------------------------
// Branch-free transcendentals (single v_exp_f32 each; correct saturation)
// ---------------------------------------------------------------------------
__device__ __forceinline__ float sigm(float x)  { return 1.f / (1.f + __expf(-x)); }
__device__ __forceinline__ float ftanh(float x) { return 1.f - 2.f / (__expf(2.f * x) + 1.f); }

// ---------------------------------------------------------------------------
// Kernel 0: pack [W_ih | W_hh]^T -> bf16 WcatT[n][k], zero h0, zero barrier
// ---------------------------------------------------------------------------
__global__ void prep_kernel(const float* __restrict__ W_ih,
                            const float* __restrict__ W_hh,
                            __bf16* __restrict__ wcatT,
                            __bf16* __restrict__ hbuf0,
                            unsigned* __restrict__ bar) {
  size_t idx = (size_t)blockIdx.x * blockDim.x + threadIdx.x;
  if (idx < (size_t)NG * KTOT) {
    int n = (int)(idx / KTOT);
    int k = (int)(idx % KTOT);
    float v = (k < Dn) ? W_ih[(size_t)n * Dn + k]
                       : W_hh[(size_t)n * Hn + (k - Dn)];
    wcatT[idx] = (__bf16)v;
  }
  if (idx < (size_t)Bn * Hn) hbuf0[idx] = (__bf16)0.0f;
  if (idx == 0) *bar = 0u;
}

// ---------------------------------------------------------------------------
// Kernel 1: attn[b,d] = softmax_d( sum_l x[b,d,l]*Wx[l] + b_attn )
// (softmax is invariant to the per-batch (h.Wh + c.Wc) shift -> time-constant)
// ---------------------------------------------------------------------------
__global__ void attn_kernel(const float* __restrict__ x,
                            const float* __restrict__ W_attn,
                            const float* __restrict__ b_attn,
                            float* __restrict__ attn) {
  __shared__ float wx[Ln];
  __shared__ float red[Dn];
  const int b = blockIdx.x;
  const int d = threadIdx.x;
  wx[d] = W_attn[2 * Hn + d];          // Wx slice of (2H+L,1)
  __syncthreads();

  const float4* r4 = (const float4*)(x + ((size_t)b * Dn + d) * Ln);
  const float4* w4 = (const float4*)wx;
  float acc = 0.f;
#pragma unroll 4
  for (int i = 0; i < Ln / 4; ++i) {
    float4 a = r4[i], w = w4[i];
    acc += a.x * w.x + a.y * w.y + a.z * w.z + a.w * w.w;
  }
  float xp = acc + b_attn[0];

  red[d] = xp; __syncthreads();
  for (int s = Dn / 2; s > 0; s >>= 1) {
    if (d < s) red[d] = fmaxf(red[d], red[d + s]);
    __syncthreads();
  }
  float mx = red[0]; __syncthreads();
  float e = __expf(xp - mx);
  red[d] = e; __syncthreads();
  for (int s = Dn / 2; s > 0; s >>= 1) {
    if (d < s) red[d] += red[d + s];
    __syncthreads();
  }
  attn[(size_t)b * Dn + d] = e / red[0];
}

// ---------------------------------------------------------------------------
// Kernel 2: ws_out[b][t][d] = attn[b,d]*x[b,d,t]  (fp32 output #1)
//           wbf[t][b][d]    = bf16 of same value  (GEMM A operand)
// 32x32 LDS transpose tiles, coalesced both sides.
// ---------------------------------------------------------------------------
__global__ void wpack_kernel(const float* __restrict__ x,
                             const float* __restrict__ attn,
                             float* __restrict__ ws_out,
                             __bf16* __restrict__ wbf) {
  __shared__ float tile[32][33];
  const int b  = blockIdx.z;
  const int t0 = blockIdx.x * 32;
  const int d0 = blockIdx.y * 32;
  const int tx = threadIdx.x & 31;      // t within tile (load) / d (store)
  const int ty = threadIdx.x >> 5;      // 0..7
#pragma unroll
  for (int i = 0; i < 4; ++i) {
    int dl = ty + i * 8;
    float a = attn[(size_t)b * Dn + d0 + dl];
    float v = x[((size_t)b * Dn + d0 + dl) * Ln + t0 + tx];
    tile[dl][tx] = a * v;
  }
  __syncthreads();
#pragma unroll
  for (int i = 0; i < 4; ++i) {
    int tl = ty + i * 8;
    float v = tile[tx][tl];
    ws_out[((size_t)b * Ln + (t0 + tl)) * Dn + d0 + tx] = v;
    wbf[((size_t)(t0 + tl) * Bn + b) * Dn + d0 + tx] = (__bf16)v;
  }
}

// ---------------------------------------------------------------------------
// Device-scope barrier (monotone ticket counter, zeroed per launch)
// ---------------------------------------------------------------------------
__device__ __forceinline__ void grid_sync(unsigned* cnt, unsigned nwg) {
  __threadfence();          // release our h writes to device scope
  __syncthreads();
  if (threadIdx.x == 0) {
    unsigned ticket = __hip_atomic_fetch_add(cnt, 1u, __ATOMIC_ACQ_REL,
                                             __HIP_MEMORY_SCOPE_AGENT);
    unsigned target = (ticket / nwg + 1u) * nwg;
    while (__hip_atomic_load(cnt, __ATOMIC_ACQUIRE,
                             __HIP_MEMORY_SCOPE_AGENT) < target) {
      __builtin_amdgcn_s_sleep(2);
    }
  }
  __threadfence();          // acquire: invalidate stale near caches
  __syncthreads();
}

// ---------------------------------------------------------------------------
// Kernel 3: persistent LSTM scan.
// Grid = (4 batch-tiles of 64) x (16 hidden-tiles of 32). 256 thr = 8 waves.
// LDS: weight slice WcatT cols {g*512 + jy*32 + j : g<4, j<32} as [128][768]
//      bf16 (192KB, loaded once), + double-buffered 64x32 bf16 A chunks.
// Wave w: mtile = w&3 (16 rows), q = w>>2 (j half). acc[g] = gate-g tile, so
// i/f/g/o for cell (b, jy*32+q*16+lane) land in one lane -> register LSTM,
// c kept in 8 regs/lane for all 256 steps. h double-buffered bf16 in global,
// one device barrier per step.
// ---------------------------------------------------------------------------
__global__ void __launch_bounds__(256, 1)
lstm_scan_kernel(const __bf16* __restrict__ wbf,
                 const __bf16* __restrict__ wcatT,
                 const float* __restrict__ b_ih,
                 const float* __restrict__ b_hh,
                 __bf16* __restrict__ hbuf,     // [2][Bn][Hn]
                 float* __restrict__ hs_out,    // [Bn][Ln][Hn]
                 unsigned* __restrict__ bar) {
  extern __shared__ char smem[];
  __bf16* Wlds  = (__bf16*)smem;                       // 128 x 768
  __bf16* Atile = (__bf16*)(smem + 128 * KTOT * 2);    // 2 x (64 x 32)

  const int tid  = threadIdx.x;
  const int lane = tid & 31, wave = tid >> 5;
  const int lo = lane & 15, hi = lane >> 4;
  const int b0 = blockIdx.x * 64;
  const int jy = blockIdx.y;

  // ---- weight slice -> LDS (once; 128 rows x 1536B, b128 copies) ----------
  for (int i = tid; i < 128 * 96; i += 256) {
    int row = i / 96, seg = i % 96;
    int col = (row >> 5) * Hn + jy * 32 + (row & 31);  // gate-major column
    ((uint4*)(Wlds + (size_t)row * KTOT))[seg] =
        ((const uint4*)(wcatT + (size_t)col * KTOT))[seg];
  }

  const int mtile = wave & 3;
  const int q     = wave >> 2;
  const int m0    = mtile * 16;
  const int hidx  = jy * 32 + q * 16 + lo;   // hidden index owned by lane

  float bias[4];
#pragma unroll
  for (int g = 0; g < 4; ++g) {
    int col = g * Hn + hidx;
    bias[g] = b_ih[col] + b_hh[col];
  }
  float creg[8];
#pragma unroll
  for (int r = 0; r < 8; ++r) creg[r] = 0.f;
  __syncthreads();

  const int arow = tid >> 2, aseg = tid & 3;  // A staging: 64 rows x 4x16B

  for (int t = 0; t < Ln; ++t) {
    const __bf16* hcur = hbuf + (size_t)(t & 1) * (Bn * Hn);
    __bf16* hnext      = hbuf + (size_t)((t + 1) & 1) * (Bn * Hn);
    v8f acc[4] = {v8f{}, v8f{}, v8f{}, v8f{}};

    // chunk source: k-chunks 0..7 from w_t, 8..23 from h
    const __bf16* srcW = wbf  + ((size_t)t * Bn + (b0 + arow)) * Dn;
    const __bf16* srcH = hcur + (size_t)(b0 + arow) * Hn;

    // prologue: stage chunk 0 into buffer 0
    ((uint4*)(Atile + arow * 32))[aseg] =
        ((const uint4*)srcW)[aseg];
    __syncthreads();

    for (int kci = 0; kci < KCH; ++kci) {
      const int kc = kci * 32;
      // prefetch next chunk into registers (latency hidden behind WMMAs)
      uint4 nxt;
      const bool have_next = (kci + 1 < KCH);
      if (have_next) {
        const int kn = kc + 32;
        const __bf16* src = (kn < Dn) ? (srcW + kn) : (srcH + (kn - Dn));
        nxt = ((const uint4*)src)[aseg];
      }

      // A fragment: lane row m0+lo; v0..3:k=8hi+0..7, v4..7:k=16+8hi+0..7
      const __bf16* Acur = Atile + (size_t)(kci & 1) * (64 * 32);
      Frag16 fa;
      const uint4* ar = (const uint4*)(Acur + (m0 + lo) * 32);
      fa.q[0] = ar[hi];
      fa.q[1] = ar[2 + hi];
#pragma unroll
      for (int g = 0; g < 4; ++g) {
        // B fragment: lane col n=g*32+q*16+lo; v0..3:k=8hi+0..7, v4..7:k=16+8hi+0..7
        Frag16 fb;
        const uint4* br = (const uint4*)(Wlds +
            (size_t)(g * 32 + q * 16 + lo) * KTOT + kc);
        fb.q[0] = br[hi];
        fb.q[1] = br[2 + hi];
        acc[g] = __builtin_amdgcn_wmma_f32_16x16x32_bf16(
            false, fa.v, false, fb.v, (short)0, acc[g], false, false);
      }

      // stage prefetched chunk into the other buffer (no readers this iter)
      if (have_next) {
        __bf16* Anext = Atile + (size_t)((kci + 1) & 1) * (64 * 32);
        ((uint4*)(Anext + arow * 32))[aseg] = nxt;
      }
      __syncthreads();
    }

    // ---- in-register LSTM epilogue: cell (b = b0+m0+r+8hi, hidx) ----------
#pragma unroll
    for (int r = 0; r < 8; ++r) {
      int b = b0 + m0 + r + 8 * hi;
      float iv = acc[0][r] + bias[0];
      float fv = acc[1][r] + bias[1];
      float gv = acc[2][r] + bias[2];
      float ov = acc[3][r] + bias[3];
      float c2 = sigm(fv) * creg[r] + sigm(iv) * ftanh(gv);
      float h2 = sigm(ov) * ftanh(c2);
      creg[r] = c2;
      hs_out[((size_t)b * Ln + t) * Hn + hidx] = h2;
      hnext[(size_t)b * Hn + hidx] = (__bf16)h2;
    }
    grid_sync(bar, gridDim.x * gridDim.y);
  }
}

// ---------------------------------------------------------------------------
extern "C" void kernel_launch(void* const* d_in, const int* in_sizes, int n_in,
                              void* d_out, int out_size, void* d_ws, size_t ws_size,
                              hipStream_t stream) {
  const float* x      = (const float*)d_in[0];   // (B,D,L)
  const float* W_attn = (const float*)d_in[1];   // (2H+L,1)
  const float* b_attn = (const float*)d_in[2];   // (1,)
  const float* W_ih   = (const float*)d_in[3];   // (4H,D)
  const float* W_hh   = (const float*)d_in[4];   // (4H,H)
  const float* b_ih   = (const float*)d_in[5];   // (4H,)
  const float* b_hh   = (const float*)d_in[6];   // (4H,)

  char* ws = (char*)d_ws;
  float*    attn  = (float*)(ws + OFF_ATTN);
  __bf16*   wcatT = (__bf16*)(ws + OFF_WCAT);
  __bf16*   wbf   = (__bf16*)(ws + OFF_WBF);
  __bf16*   hbuf  = (__bf16*)(ws + OFF_HBUF);
  unsigned* bar   = (unsigned*)(ws + OFF_BAR);

  float* ws_out = (float*)d_out;                       // (B,L,D)
  float* hs_out = (float*)d_out + (size_t)Bn * Ln * Dn;// (B,L,H)

  // 0) weight pack + state init (must rerun every launch; ws is poisoned)
  {
    int total = NG * KTOT;
    prep_kernel<<<(total + 255) / 256, 256, 0, stream>>>(W_ih, W_hh, wcatT,
                                                         hbuf, bar);
  }
  // 1) time-invariant attention weights
  attn_kernel<<<Bn, Dn, 0, stream>>>(x, W_attn, b_attn, attn);
  // 2) ws output + bf16 A operand (d<->t transpose)
  wpack_kernel<<<dim3(Ln / 32, Dn / 32, Bn), 256, 0, stream>>>(x, attn,
                                                               ws_out, wbf);
  // 3) persistent fused GEMM+LSTM scan (192KB weights + 2x4KB staging in LDS)
  size_t lds_bytes = (size_t)128 * KTOT * 2 + 2 * 64 * 32 * 2;
  lstm_scan_kernel<<<dim3(4, 16), 256, lds_bytes, stream>>>(
      wbf, wcatT, b_ih, b_hh, hbuf, hs_out, bar);
}
